// ActionHead_84567906058924
// MI455X (gfx1250) — matrix operand order repaired and checked by reading, not compiled, not run on hardware
//
#include <hip/hip_runtime.h>
#include <hip/hip_bf16.h>
#include <math.h>

// ---------------- problem constants (match reference) ----------------
#define B_      64
#define PTS_    4096
#define N_      (B_ * PTS_)       // 262144 rows
#define H_      256               // hidden dim
#define OUT_    218               // 72*3 + 1 + 1
#define POOLC_  (OUT_ - 1)        // 217 pooled columns
#define LEAK_   0.02f

// ---------------- tiling ----------------
#define WGROWS  64                // rows per workgroup
#define LDA     264               // bf16 row stride in LDS (256 + 8 pad, even)
#define NBLK    (N_ / WGROWS)     // 4096 workgroups over rows
#define CT1_    16                // GEMM1 column tiles (256/16)
#define CT2_    14                // GEMM2 column tiles (224 >= 218)

// packed-B sizes (bf16 pairs laid out in WMMA B-fragment order)
#define W1P_PAIRS (CT1_ * 8 * 32 * 8)   // 32768 u32-pairs = 65536 bf16
#define W2P_PAIRS (CT2_ * 8 * 32 * 8)   // 28672 u32-pairs = 57344 bf16

// workspace layout (float units)
#define WS_LOGITS 0
#define WS_M      (N_)
#define WS_POOLED (N_ + 64)
#define WS_DENOM  (N_ + 64 + B_ * POOLC_)
#define WS_W1P    (N_ + 64 + B_ * POOLC_ + 64)     // 276160, 32B-aligned
#define WS_W2P    (WS_W1P + W1P_PAIRS)             // W1P = 32768 floats worth

typedef __attribute__((ext_vector_type(16))) __bf16 v16bf;
typedef __attribute__((ext_vector_type(8)))  float  v8f;

// ---------------------------------------------------------------------
// One-time weight pack: f32 row-major W -> bf16 in B-fragment order.
// Fragment (ct,kt): lane -> col = ct*16+(lane&15); lanes 0-15 hold
// K=kt*32+0..15, lanes 16-31 hold K=kt*32+16..31, two K per VGPR slot v.
// Linear pair index = ((ct*8+kt)*32+lane)*8+v  -> 32B contiguous per lane.
// ---------------------------------------------------------------------
__global__ __launch_bounds__(256) void k_pack(const float* __restrict__ W1,
                                              const float* __restrict__ W2,
                                              __bf16* __restrict__ W1P,
                                              __bf16* __restrict__ W2P) {
    int i = blockIdx.x * 256 + threadIdx.x;
    if (i < W1P_PAIRS) {
        int v    = i & 7;
        int lane = (i >> 3) & 31;
        int kt   = (i >> 8) & 7;
        int ct   = i >> 11;
        int col  = ct * 16 + (lane & 15);
        int k    = kt * 32 + (lane >> 4) * 16 + 2 * v;
        W1P[2 * i]     = (__bf16)W1[(size_t)k * H_ + col];
        W1P[2 * i + 1] = (__bf16)W1[(size_t)(k + 1) * H_ + col];
    } else {
        int j = i - W1P_PAIRS;
        if (j < W2P_PAIRS) {
            int v    = j & 7;
            int lane = (j >> 3) & 31;
            int kt   = (j >> 8) & 7;
            int ct   = j >> 11;
            int col  = ct * 16 + (lane & 15);
            int k    = kt * 32 + (lane >> 4) * 16 + 2 * v;
            bool ok  = (col < OUT_);
            float f0 = ok ? W2[(size_t)k * OUT_ + col]       : 0.f;
            float f1 = ok ? W2[(size_t)(k + 1) * OUT_ + col] : 0.f;
            W2P[2 * j]     = (__bf16)f0;
            W2P[2 * j + 1] = (__bf16)f1;
        }
    }
}

// GEMM1 body is written inline in both consumer kernels so every LDS
// access is a direct, in-scope __shared__ reference (32-bit AS3 addressing,
// no generic-pointer null checks).  Layout facts used below:
//  A frag (16x32 bf16): lane&15 = row-in-tile; half = lane>>4;
//    a[0..7]  = Abf[row*LDA + kt*32 + half*16 + 0..7]
//    a[8..15] = Abf[row*LDA + kt*32 + half*16 + 8..15]   (two 16B ds_load_b128)
//  C/D (16x16 f32): VGPR r -> row rt*16 + r + 8*half, col = ct*16 + (lane&15)
#define GEMM1_STAGE_AND_COMPUTE()                                            \
    {                                                                        \
        const float4* srcA = (const float4*)(pe + (size_t)n0 * H_);          \
        _Pragma("unroll 4")                                                  \
        for (int p = tid; p < WGROWS * (H_ / 4); p += 256) {                 \
            float4 x = srcA[p];                                              \
            int row = p >> 6;                                                \
            int c4  = (p & 63) * 4;                                          \
            Abf[row * LDA + c4 + 0] = (__bf16)x.x;                           \
            Abf[row * LDA + c4 + 1] = (__bf16)x.y;                           \
            Abf[row * LDA + c4 + 2] = (__bf16)x.z;                           \
            Abf[row * LDA + c4 + 3] = (__bf16)x.w;                           \
        }                                                                    \
    }                                                                        \
    __syncthreads();                                                         \
    {                                                                        \
        const v16bf* W1Pv = (const v16bf*)W1P;                               \
        const v8f vzero = {0.f,0.f,0.f,0.f,0.f,0.f,0.f,0.f};                 \
        v8f acc[2][4];                                                       \
        _Pragma("unroll")                                                    \
        for (int i_ = 0; i_ < 2; ++i_)                                       \
            _Pragma("unroll")                                                \
            for (int rt = 0; rt < 4; ++rt) acc[i_][rt] = vzero;              \
        const int ct0 = wave, ct1 = wave + 8;                                \
        _Pragma("unroll 1")                                                  \
        for (int kt = 0; kt < 8; ++kt) {                                     \
            v16bf b0 = W1Pv[(ct0 * 8 + kt) * 32 + lane];                     \
            v16bf b1v = W1Pv[(ct1 * 8 + kt) * 32 + lane];                    \
            _Pragma("unroll")                                                \
            for (int rt = 0; rt < 4; ++rt) {                                 \
                int abase = (rt * 16 + (lane & 15)) * LDA + kt * 32 + half*16;\
                v16bf a;                                                     \
                _Pragma("unroll")                                            \
                for (int e = 0; e < 8; ++e) a[e] = Abf[abase + e];           \
                _Pragma("unroll")                                            \
                for (int e = 0; e < 8; ++e) a[8 + e] = Abf[abase + 8 + e];   \
                acc[0][rt] = __builtin_amdgcn_wmma_f32_16x16x32_bf16(        \
                    false, a, false, b0, (short)0, acc[0][rt], false, false);\
                acc[1][rt] = __builtin_amdgcn_wmma_f32_16x16x32_bf16(        \
                    false, a, false, b1v, (short)0, acc[1][rt], false,false);\
            }                                                                \
        }                                                                    \
        __syncthreads();                                                     \
        _Pragma("unroll")                                                    \
        for (int i_ = 0; i_ < 2; ++i_) {                                     \
            int col = (wave + 8 * i_) * 16 + (lane & 15);                    \
            float bi = b1[col];                                              \
            _Pragma("unroll")                                                \
            for (int rt = 0; rt < 4; ++rt)                                   \
                _Pragma("unroll")                                            \
                for (int r = 0; r < 8; ++r) {                                \
                    float hv = acc[i_][rt][r] + bi;                          \
                    hv = (hv >= 0.f) ? hv : LEAK_ * hv;                      \
                    Abf[(rt * 16 + r + 8 * half) * LDA + col] = (__bf16)hv;  \
                }                                                            \
        }                                                                    \
    }                                                                        \
    __syncthreads();

// ---------------------------------------------------------------------
// Pass 1: logits[n] = h(n) . W2[:,0] + b2[0]
// ---------------------------------------------------------------------
__global__ __launch_bounds__(256) void k_logits(const float* __restrict__ pe,
                                                const __bf16* __restrict__ W1P,
                                                const float* __restrict__ b1,
                                                const float* __restrict__ W2,
                                                const float* __restrict__ b2,
                                                float* __restrict__ logits) {
    __shared__ __bf16 Abf[WGROWS * LDA];
    __shared__ float  w2c0[H_];
    __shared__ float  partial[4][WGROWS];

    const int n0   = blockIdx.x * WGROWS;
    const int tid  = threadIdx.x;
    const int lane = tid & 31;
    const int wave = tid >> 5;
    const int half = lane >> 4;

    // stage W2 column 0 while A tile streams in
    if (tid < H_) w2c0[tid] = W2[(size_t)tid * OUT_];

    GEMM1_STAGE_AND_COMPUTE()

    const int row = tid & 63;
    const int q   = tid >> 6;        // quarter of K
    float s = 0.f;
#pragma unroll 8
    for (int k = q * 64; k < q * 64 + 64; ++k)
        s += (float)Abf[row * LDA + k] * w2c0[k];
    partial[q][row] = s;
    __syncthreads();
    if (tid < WGROWS) {
        float l = b2[0] + partial[0][tid] + partial[1][tid] +
                  partial[2][tid] + partial[3][tid];
        logits[n0 + tid] = l;
    }
}

// ---------------------------------------------------------------------
// Segment max: one block per batch, 4096 logits each
// ---------------------------------------------------------------------
__global__ __launch_bounds__(256) void k_segmax(const float* __restrict__ logits,
                                                float* __restrict__ m) {
    __shared__ float red[256];
    const int b = blockIdx.x;
    float mx = -3.0e38f;
    for (int i = threadIdx.x; i < PTS_; i += 256)
        mx = fmaxf(mx, logits[b * PTS_ + i]);
    red[threadIdx.x] = mx;
    __syncthreads();
    for (int s = 128; s > 0; s >>= 1) {
        if (threadIdx.x < s)
            red[threadIdx.x] = fmaxf(red[threadIdx.x], red[threadIdx.x + s]);
        __syncthreads();
    }
    if (threadIdx.x == 0) m[b] = red[0];
}

// ---------------------------------------------------------------------
// Pass 2: recompute h, full GEMM2, accumulate softmax-weighted pooling
// ---------------------------------------------------------------------
__global__ __launch_bounds__(256) void k_main(const float* __restrict__ pe,
                                              const __bf16* __restrict__ W1P,
                                              const float* __restrict__ b1,
                                              const __bf16* __restrict__ W2P,
                                              const float* __restrict__ b2,
                                              const float* __restrict__ logits,
                                              const float* __restrict__ m,
                                              float* __restrict__ pooled,
                                              float* __restrict__ denom) {
    __shared__ __bf16 Abf[WGROWS * LDA];
    __shared__ float  eL[WGROWS];

    const int n0   = blockIdx.x * WGROWS;
    const int b    = n0 / PTS_;          // tile lies within one segment
    const int tid  = threadIdx.x;
    const int lane = tid & 31;
    const int wave = tid >> 5;
    const int half = lane >> 4;

    GEMM1_STAGE_AND_COMPUTE()

    if (tid < WGROWS) eL[tid] = __expf(logits[n0 + tid] - m[b]);
    __syncthreads();

    // denominator contribution (two waves reduce 64 values)
    if (tid < WGROWS) {
        float v = eL[tid];
#pragma unroll
        for (int off = 16; off > 0; off >>= 1) v += __shfl_down(v, off, 32);
        if (lane == 0) atomicAdd(&denom[b], v);
    }

    // GEMM2: 14 column tiles cover 224 >= 218 cols; wave-uniform loop
    const v16bf* W2Pv = (const v16bf*)W2P;
    for (int ct = wave; ct < CT2_; ct += 8) {
        const v8f vzero = {0.f,0.f,0.f,0.f,0.f,0.f,0.f,0.f};
        v8f acc[4];
#pragma unroll
        for (int rt = 0; rt < 4; ++rt) acc[rt] = vzero;

#pragma unroll 1
        for (int kt = 0; kt < 8; ++kt) {
            v16bf bf = W2Pv[(ct * 8 + kt) * 32 + lane];
#pragma unroll
            for (int rt = 0; rt < 4; ++rt) {
                int abase = (rt * 16 + (lane & 15)) * LDA + kt * 32 + half * 16;
                v16bf a;
#pragma unroll
                for (int e = 0; e < 8; ++e) a[e] = Abf[abase + e];
#pragma unroll
                for (int e = 0; e < 8; ++e) a[8 + e] = Abf[abase + 8 + e];
                acc[rt] = __builtin_amdgcn_wmma_f32_16x16x32_bf16(
                    false, a, false, bf, (short)0, acc[rt], false, false);
            }
        }

        const int col  = ct * 16 + (lane & 15);
        const int colc = (col < OUT_) ? col : (OUT_ - 1);
        const float bias = b2[colc];
        float csum = 0.f;
#pragma unroll
        for (int rt = 0; rt < 4; ++rt) {
#pragma unroll
            for (int r = 0; r < 8; ++r) {
                int row  = rt * 16 + r + 8 * half;
                float ae = acc[rt][r] + bias;
                csum += eL[row] * ae;
            }
        }
        csum += __shfl_xor(csum, 16, 32);   // combine row-halves
        if (half == 0 && col >= 1 && col < OUT_)
            atomicAdd(&pooled[b * POOLC_ + (col - 1)], csum);
    }
}

// ---------------------------------------------------------------------
// Init / finalize
// ---------------------------------------------------------------------
__global__ void k_init(float* __restrict__ pooled, float* __restrict__ denom) {
    int i = blockIdx.x * 256 + threadIdx.x;
    if (i < B_ * POOLC_) pooled[i] = 0.f;
    if (i < B_) denom[i] = 0.f;
}

__global__ void k_final(const float* __restrict__ pooled,
                        const float* __restrict__ denom,
                        float* __restrict__ out) {
    int i = blockIdx.x * 256 + threadIdx.x;
    if (i >= B_ * POOLC_) return;
    int b = i / POOLC_;
    int j = i % POOLC_;
    float v = pooled[i] / denom[b];
    if (j < 216) out[b * 216 + j] = v;      // xr, flat (B,72,3)
    else         out[B_ * 216 + b] = v;     // xo
}

// ---------------------------------------------------------------------
extern "C" void kernel_launch(void* const* d_in, const int* in_sizes, int n_in,
                              void* d_out, int out_size, void* d_ws, size_t ws_size,
                              hipStream_t stream) {
    const float* pe = (const float*)d_in[0];   // point_embeds (N, H)
    const float* W1 = (const float*)d_in[1];   // (H, H)
    const float* b1 = (const float*)d_in[2];   // (H,)
    const float* W2 = (const float*)d_in[3];   // (H, OUT)
    const float* b2 = (const float*)d_in[4];   // (OUT,)
    // d_in[5] = npoints_in_batch (constant PTS), d_in[6] = noise_step (unused by ref)

    float* ws     = (float*)d_ws;
    float*  logits = ws + WS_LOGITS;
    float*  mseg   = ws + WS_M;
    float*  pooled = ws + WS_POOLED;
    float*  denom  = ws + WS_DENOM;
    __bf16* W1P    = (__bf16*)(ws + WS_W1P);
    __bf16* W2P    = (__bf16*)(ws + WS_W2P);

    float* out = (float*)d_out;

    const int initBlocks = (B_ * POOLC_ + 255) / 256;
    const int packBlocks = (W1P_PAIRS + W2P_PAIRS + 255) / 256;

    k_pack  <<<packBlocks, 256, 0, stream>>>(W1, W2, W1P, W2P);
    k_init  <<<initBlocks, 256, 0, stream>>>(pooled, denom);
    k_logits<<<NBLK,       256, 0, stream>>>(pe, W1P, b1, W2, b2, logits);
    k_segmax<<<B_,         256, 0, stream>>>(logits, mseg);
    k_main  <<<NBLK,       256, 0, stream>>>(pe, W1P, b1, W2P, b2, logits, mseg,
                                             pooled, denom);
    k_final <<<initBlocks, 256, 0, stream>>>(pooled, denom, out);
}
